// dblp_hGAT_21534966022331
// MI455X (gfx1250) — matrix operand order repaired and checked by reading, not compiled, not run on hardware
//
#include <hip/hip_runtime.h>
#include <hip/hip_bf16.h>
#include <math.h>

#define NN      50000
#define NN0     25000
#define EE      400000
#define D_IN    256
#define D_H     64

typedef __attribute__((ext_vector_type(16))) _Float16 v16h;
typedef __attribute__((ext_vector_type(8)))  _Float16 v8h;
typedef __attribute__((ext_vector_type(4)))  _Float16 v4h;
typedef __attribute__((ext_vector_type(8)))  float    v8f;

// ---------------- utility kernels ----------------

__global__ void fill_kernel(float* __restrict__ p, float v, long long n) {
    long long t = (long long)blockIdx.x * blockDim.x + threadIdx.x;
    if (t < n) p[t] = v;
}

// f32 -> f16 bulk convert (n must be a multiple of 4; all our sizes are)
__global__ void cvt_f32_f16_kernel(const float* __restrict__ in, _Float16* __restrict__ out,
                                   long long n4) {
    long long t = (long long)blockIdx.x * blockDim.x + threadIdx.x;
    if (t >= n4) return;
    const float4 v = *(const float4*)(in + t * 4);
    v4h o;
    o[0] = (_Float16)v.x; o[1] = (_Float16)v.y; o[2] = (_Float16)v.z; o[3] = (_Float16)v.w;
    *(v4h*)(out + t * 4) = o;
}

// weights: B (K x Nc, f32, row-major) -> Bt (Nc x K, f16)
__global__ void cvt_w_transpose_kernel(const float* __restrict__ B, _Float16* __restrict__ Bt,
                                       int K, int Nc) {
    int t = blockIdx.x * blockDim.x + threadIdx.x;
    if (t >= K * Nc) return;
    const int k   = t % K;
    const int col = t / K;
    Bt[(size_t)col * K + k] = (_Float16)B[(size_t)k * Nc + col];
}

__device__ __forceinline__ void atomic_max_f32(float* addr, float val) {
    unsigned int* ua = (unsigned int*)addr;
    unsigned int old = *ua;
    while (__uint_as_float(old) < val) {
        unsigned int assumed = old;
        old = atomicCAS(ua, assumed, __float_as_uint(val));
        if (old == assumed) break;
    }
}

// ---------------- WMMA GEMM with LDS-staged weights ----------------
// C[M,Nc] = A[M,K](f16) @ Bt[Nc,K](f16, pre-transposed) (+bias)
// Workgroup = 8 waves = 128 M-rows x 64 N-cols. The 64xK weight tile
// (contiguous span of Bt) is staged once into LDS cooperatively; each wave
// then computes a 16x64 tile: per 32-wide K-step one A fragment (2x b128
// global) feeds 4 WMMAs whose B fragments come from LDS (ds_load_b128).
// K in {64, 256} (templated -> fully unrolled k-loop); Nc multiple of 64.

template <int K>
__global__ void __launch_bounds__(256)
gemm_wmma_lds_kernel(const _Float16* __restrict__ A,
                     const _Float16* __restrict__ Bt,
                     const float* __restrict__ bias,
                     float* __restrict__ C,
                     int M, int Nc) {
    __shared__ __align__(16) _Float16 sB[64 * 256];

    const int tid  = threadIdx.x;
    const int nblk = Nc >> 6;
    const int bn   = blockIdx.x % nblk;
    const int bm   = blockIdx.x / nblk;
    const int n_base = bn << 6;

    // cooperative LDS fill: 64 consecutive Bt rows = one contiguous span
    {
        const _Float16* bsrc = Bt + (size_t)n_base * K;
        const int nchunks = (64 * K) / 8;            // v8h (16B) chunks
        #pragma unroll
        for (int c0 = 0; c0 < (64 * 256) / 8; c0 += 256) {
            const int c = c0 + tid;
            if (c < nchunks) *(v8h*)(sB + c * 8) = *(const v8h*)(bsrc + (size_t)c * 8);
        }
    }
    __syncthreads();

    const int lane    = tid & 31;
    const int wave    = tid >> 5;
    const int m_base  = (bm * 8 + wave) << 4;
    if (m_base >= M) return;                         // wave-uniform

    const int half_id = lane >> 4;                   // K-half selector (ISA fragment layout)
    const int l15     = lane & 15;

    int arow = m_base + l15;
    if (arow >= M) arow = M - 1;                     // clamp; stores guarded below
    const _Float16* ap = A + (size_t)arow * K;

    v8f acc[4] = {v8f{}, v8f{}, v8f{}, v8f{}};

    #pragma unroll
    for (int k0 = 0; k0 < K; k0 += 32) {
        // A fragment 16x32: lanes 0-15 -> K {0..7,16..23}, lanes 16-31 -> {8..15,24..31}
        const v8h alo = *(const v8h*)(ap + k0 + half_id * 8);
        const v8h ahi = *(const v8h*)(ap + k0 + 16 + half_id * 8);
        const v16h a = __builtin_shufflevector(alo, ahi,
                           0, 1, 2, 3, 4, 5, 6, 7, 8, 9, 10, 11, 12, 13, 14, 15);
        #pragma unroll
        for (int s = 0; s < 4; ++s) {
            // B fragment 32x16 from LDS: lane holds col, elements K = {half*8+0..7, 16+half*8+0..7}
            const _Float16* bp = sB + (size_t)(s * 16 + l15) * K + k0 + half_id * 8;
            const v8h blo = *(const v8h*)bp;
            const v8h bhi = *(const v8h*)(bp + 16);
            const v16h b = __builtin_shufflevector(blo, bhi,
                               0, 1, 2, 3, 4, 5, 6, 7, 8, 9, 10, 11, 12, 13, 14, 15);
            acc[s] = __builtin_amdgcn_wmma_f32_16x16x32_f16(false, a, false, b,
                                                            (short)0, acc[s], false, false);
        }
    }

    // C/D layout: element v at lane l -> row m_base + (l>>4)*8 + v, col (l&15)
    const bool full = (m_base + 16) <= M;            // wave-uniform
    #pragma unroll
    for (int s = 0; s < 4; ++s) {
        const int col = n_base + s * 16 + l15;
        const float bv = bias ? bias[col] : 0.0f;
        float* cp = C + (size_t)(m_base + half_id * 8) * Nc + col;
        if (full) {
            #pragma unroll
            for (int v = 0; v < 8; ++v) cp[(size_t)v * Nc] = acc[s][v] + bv;
        } else {
            #pragma unroll
            for (int v = 0; v < 8; ++v)
                if (m_base + half_id * 8 + v < M) cp[(size_t)v * Nc] = acc[s][v] + bv;
        }
    }
}

// ---------------- attention scores ----------------

__global__ void attn_scores_kernel(const float* __restrict__ z,
                                   const float* __restrict__ al,
                                   const float* __restrict__ ar,
                                   float* __restrict__ el,
                                   float* __restrict__ er,
                                   int H) {
    int t = blockIdx.x * blockDim.x + threadIdx.x;
    if (t >= NN * H) return;
    const int h    = t % H;
    const int node = t / H;
    const float* zp  = z  + (size_t)node * H * D_H + (size_t)h * D_H;
    const float* alp = al + h * D_H;
    const float* arp = ar + h * D_H;
    float sl = 0.f, sr = 0.f;
    #pragma unroll 8
    for (int d = 0; d < D_H; ++d) {
        const float zv = zp[d];
        sl += zv * alp[d];
        sr += zv * arp[d];
    }
    el[t] = sl;
    er[t] = sr;
}

__device__ __forceinline__ float leaky02(float x) { return x > 0.f ? x : 0.2f * x; }

// ---------------- edge softmax passes ----------------

__global__ void edge_max_kernel(const float* __restrict__ el, const float* __restrict__ er,
                                const int* __restrict__ src, const int* __restrict__ dst,
                                float* __restrict__ dmax, int H) {
    int t = blockIdx.x * blockDim.x + threadIdx.x;
    if (t >= EE * H) return;
    const int h = t % H;
    const int i = t / H;
    const float e = leaky02(el[src[i] * H + h] + er[dst[i] * H + h]);
    atomic_max_f32(&dmax[dst[i] * H + h], e);
}

__global__ void edge_expsum_kernel(const float* __restrict__ el, const float* __restrict__ er,
                                   const int* __restrict__ src, const int* __restrict__ dst,
                                   const float* __restrict__ dmax,
                                   float* __restrict__ dsum, float* __restrict__ eex, int H) {
    int t = blockIdx.x * blockDim.x + threadIdx.x;
    if (t >= EE * H) return;
    const int h  = t % H;
    const int i  = t / H;
    const int dd = dst[i];
    const float e  = leaky02(el[src[i] * H + h] + er[dd * H + h]);
    const float ex = __expf(e - dmax[dd * H + h]);
    eex[t] = ex;
    atomicAdd(&dsum[dd * H + h], ex);
}

__global__ void edge_aggregate_kernel(const float* __restrict__ z,
                                      const float* __restrict__ eex, const float* __restrict__ dsum,
                                      const int* __restrict__ src, const int* __restrict__ dst,
                                      float* __restrict__ out, int H) {
    long long t = (long long)blockIdx.x * blockDim.x + threadIdx.x;
    if (t >= (long long)EE * H * D_H) return;
    const int d = (int)(t % D_H);
    const int h = (int)((t / D_H) % H);
    const int i = (int)(t / ((long long)D_H * H));
    const int s  = src[i];
    const int dd = dst[i];
    const float a = eex[i * H + h] / dsum[dd * H + h];
    atomicAdd(&out[(size_t)dd * H * D_H + (size_t)h * D_H + d],
              a * z[(size_t)s * H * D_H + (size_t)h * D_H + d]);
}

__global__ void bias_act_kernel(float* __restrict__ out, const float* __restrict__ bias,
                                int width, int use_elu) {
    long long t = (long long)blockIdx.x * blockDim.x + threadIdx.x;
    if (t >= (long long)NN * width) return;
    const int c = (int)(t % width);
    float x = out[t] + bias[c];
    if (use_elu) x = x > 0.f ? x : (__expf(x) - 1.f);
    out[t] = x;
}

// ---------------- final edge-type message passing ----------------

__global__ void twohop_kernel(const float* __restrict__ h, const int* __restrict__ ef,
                              const int* __restrict__ src, const int* __restrict__ dst,
                              float* __restrict__ ft) {
    int t = blockIdx.x * blockDim.x + threadIdx.x;
    if (t >= EE * D_H) return;
    const int d  = t % D_H;
    const int i  = t / D_H;
    const int et = ef[i];
    const float cc = 1.f + (float)(et == 0) + (float)(et == 4) + (float)(et == 5);
    atomicAdd(&ft[(size_t)dst[i] * D_H + d], cc * h[(size_t)src[i] * D_H + d]);
}

__global__ void final_hop_kernel(const float* __restrict__ ft, const int* __restrict__ ef,
                                 const int* __restrict__ src, const int* __restrict__ dst,
                                 float* __restrict__ res) {
    int t = blockIdx.x * blockDim.x + threadIdx.x;
    if (t >= EE * D_H) return;
    const int d = t % D_H;
    const int i = t / D_H;
    if (ef[i] == 3)
        atomicAdd(&res[(size_t)dst[i] * D_H + d], ft[(size_t)src[i] * D_H + d]);
}

// ---------------- host orchestration ----------------

static inline unsigned cdivu(long long a, long long b) { return (unsigned)((a + b - 1) / b); }

static void launch_gemm(const _Float16* A, const _Float16* Bt, const float* bias, float* C,
                        int M, int K, int Nc, hipStream_t stream) {
    const int nblk    = Nc / 64;
    const int mblk    = (M + 15) / 16;
    const int mchunks = (mblk + 7) / 8;            // 8 waves (16 rows each) per block
    const int blocks  = mchunks * nblk;
    if (K == 64) {
        gemm_wmma_lds_kernel<64><<<blocks, 256, 0, stream>>>(A, Bt, bias, C, M, Nc);
    } else {
        gemm_wmma_lds_kernel<256><<<blocks, 256, 0, stream>>>(A, Bt, bias, C, M, Nc);
    }
}

static void cvt_act(const float* in, _Float16* out, long long n, hipStream_t stream) {
    cvt_f32_f16_kernel<<<cdivu(n / 4, 256), 256, 0, stream>>>(in, out, n / 4);
}

static void cvt_wt(const float* W, _Float16* Bt, int K, int Nc, hipStream_t stream) {
    cvt_w_transpose_kernel<<<cdivu((long long)K * Nc, 256), 256, 0, stream>>>(W, Bt, K, Nc);
}

// z = hin @ W, then edge-softmax attention aggregation with bias/activation.
static void run_gat_layer(const float* hin, int Kin,
                          const float* W, const float* al, const float* ar, const float* bias,
                          int H, const int* src, const int* dst,
                          _Float16* hA, _Float16* wBt,
                          float* z, float* out,
                          float* el, float* er, float* dmax, float* dsum, float* eex,
                          int use_elu, hipStream_t stream) {
    const int Nc = H * D_H;
    cvt_act(hin, hA, (long long)NN * Kin, stream);
    cvt_wt(W, wBt, Kin, Nc, stream);
    launch_gemm(hA, wBt, nullptr, z, NN, Kin, Nc, stream);
    attn_scores_kernel<<<cdivu((long long)NN * H, 256), 256, 0, stream>>>(z, al, ar, el, er, H);
    fill_kernel<<<cdivu((long long)NN * H, 256), 256, 0, stream>>>(dmax, -1e30f, (long long)NN * H);
    fill_kernel<<<cdivu((long long)NN * H, 256), 256, 0, stream>>>(dsum, 0.f, (long long)NN * H);
    fill_kernel<<<cdivu((long long)NN * Nc, 256), 256, 0, stream>>>(out, 0.f, (long long)NN * Nc);
    edge_max_kernel<<<cdivu((long long)EE * H, 256), 256, 0, stream>>>(el, er, src, dst, dmax, H);
    edge_expsum_kernel<<<cdivu((long long)EE * H, 256), 256, 0, stream>>>(el, er, src, dst, dmax, dsum, eex, H);
    edge_aggregate_kernel<<<cdivu((long long)EE * H * D_H, 256), 256, 0, stream>>>(z, eex, dsum, src, dst, out, H);
    bias_act_kernel<<<cdivu((long long)NN * Nc, 256), 256, 0, stream>>>(out, bias, Nc, use_elu);
}

extern "C" void kernel_launch(void* const* d_in, const int* in_sizes, int n_in,
                              void* d_out, int out_size, void* d_ws, size_t ws_size,
                              hipStream_t stream) {
    const float* feat0 = (const float*)d_in[0];
    const float* feat1 = (const float*)d_in[1];
    const float* fc0_w = (const float*)d_in[2];
    const float* fc0_b = (const float*)d_in[3];
    const float* fc1_w = (const float*)d_in[4];
    const float* fc1_b = (const float*)d_in[5];
    const float* g0_w  = (const float*)d_in[6];
    const float* g0_al = (const float*)d_in[7];
    const float* g0_ar = (const float*)d_in[8];
    const float* g0_b  = (const float*)d_in[9];
    const float* g1_w  = (const float*)d_in[10];
    const float* g1_al = (const float*)d_in[11];
    const float* g1_ar = (const float*)d_in[12];
    const float* g1_b  = (const float*)d_in[13];
    const float* g2_w  = (const float*)d_in[14];
    const float* g2_al = (const float*)d_in[15];
    const float* g2_ar = (const float*)d_in[16];
    const float* g2_b  = (const float*)d_in[17];
    const int*   e_ft  = (const int*)d_in[18];
    const int*   src   = (const int*)d_in[19];
    const int*   dst   = (const int*)d_in[20];
    float*       res   = (float*)d_out;

    // workspace layout (float units; f16 regions sized in float-equivalents)
    float* ws = (float*)d_ws;
    size_t off = 0;
    float* buf0 = ws + off; off += (size_t)NN * 256;          // f32 node buffer
    float* buf1 = ws + off; off += (size_t)NN * 256;          // f32 node buffer
    _Float16* hA  = (_Float16*)(ws + off); off += (size_t)NN * 256 / 2;   // f16 activations
    _Float16* wBt = (_Float16*)(ws + off); off += (size_t)(256 * 256) / 2; // f16 weights^T
    float* el   = ws + off; off += (size_t)NN * 4;
    float* er   = ws + off; off += (size_t)NN * 4;
    float* dmax = ws + off; off += (size_t)NN * 4;
    float* dsum = ws + off; off += (size_t)NN * 4;
    float* eex  = ws + off; off += (size_t)EE * 4;
    float* ft   = ws + off; off += (size_t)NN * D_H;
    (void)ws_size; (void)in_sizes; (void)n_in; (void)out_size;

    // 1) input projections -> buf0 (NN x 64)
    cvt_act(feat0, hA, (long long)NN0 * D_IN, stream);
    cvt_act(feat1, hA + (size_t)NN0 * D_IN, (long long)NN0 * D_IN, stream);
    cvt_wt(fc0_w, wBt, D_IN, D_H, stream);
    launch_gemm(hA, wBt, fc0_b, buf0, NN0, D_IN, D_H, stream);
    cvt_wt(fc1_w, wBt, D_IN, D_H, stream);
    launch_gemm(hA + (size_t)NN0 * D_IN, wBt, fc1_b, buf0 + (size_t)NN0 * D_H,
                NN0, D_IN, D_H, stream);

    // 2) GAT layer 0: in buf0 (K=64); z overwrites buf0 (input lives in hA); out buf1; ELU
    run_gat_layer(buf0, 64,  g0_w, g0_al, g0_ar, g0_b, 4, src, dst,
                  hA, wBt, buf0, buf1, el, er, dmax, dsum, eex, 1, stream);

    // 3) GAT layer 1: in buf1 (K=256); z overwrites buf1; out buf0; ELU
    run_gat_layer(buf1, 256, g1_w, g1_al, g1_ar, g1_b, 4, src, dst,
                  hA, wBt, buf1, buf0, el, er, dmax, dsum, eex, 1, stream);

    // 4) GAT layer 2: in buf0 (K=256); z overwrites buf0 (Nx64); out buf1; no act; H=1
    run_gat_layer(buf0, 256, g2_w, g2_al, g2_ar, g2_b, 1, src, dst,
                  hA, wBt, buf0, buf1, el, er, dmax, dsum, eex, 0, stream);

    // 5) edge-type two-hop message passing on h = buf1 (NN x 64)
    fill_kernel<<<cdivu((long long)NN * D_H, 256), 256, 0, stream>>>(ft,  0.f, (long long)NN * D_H);
    fill_kernel<<<cdivu((long long)NN * D_H, 256), 256, 0, stream>>>(res, 0.f, (long long)NN * D_H);
    twohop_kernel   <<<cdivu((long long)EE * D_H, 256), 256, 0, stream>>>(buf1, e_ft, src, dst, ft);
    final_hop_kernel<<<cdivu((long long)EE * D_H, 256), 256, 0, stream>>>(ft,   e_ft, src, dst, res);
}
